// TreeLSTM_2602750181891
// MI455X (gfx1250) — matrix operand order repaired and checked by reading, not compiled, not run on hardware
//
#include <hip/hip_runtime.h>
#include <math.h>

// ---------------------------------------------------------------------------
// TreeLSTM on gfx1250: bf16 WMMA (v_wmma_f32_16x16x32_bf16), fp32 accumulate,
// async global->LDS staging (global_load_async_to_lds_b128 + s_wait_asynccnt).
// Constants: B=32768, DEPTH=3, NLEAF=262144, V=32000, X=256, H=128, C=5.
// ---------------------------------------------------------------------------

typedef __attribute__((ext_vector_type(16))) __bf16 v16bf;
typedef __attribute__((ext_vector_type(8)))  float  v8f;
typedef __attribute__((ext_vector_type(4)))  int    vi4;

union ABPack { v16bf v; vi4 q[2]; };

#define XDIM 256       // K of the GEMMs (input / h_cat width)
#define HDIM 128       // hidden width
#define LDSPITCH 264   // bf16 elements per LDS row (256 + 8 pad -> bank spread)

__device__ __forceinline__ unsigned short f2bf(float f) {
  unsigned int u = __float_as_uint(f);
  u += 0x7FFFu + ((u >> 16) & 1u);           // round-to-nearest-even
  return (unsigned short)(u >> 16);
}

__device__ __forceinline__ float bf2f(unsigned short b) {
  return __uint_as_float(((unsigned int)b) << 16);
}

__device__ __forceinline__ float sigf(float x) {
  return 1.0f / (1.0f + __expf(-x));
}

// Async copy of 16 bytes global -> LDS (per lane), tracked by ASYNCcnt.
__device__ __forceinline__ void async_b128(unsigned lds_off, const void* gptr) {
  unsigned long long ga = (unsigned long long)gptr;
  asm volatile("global_load_async_to_lds_b128 %0, %1, off"
               :: "v"(lds_off), "v"(ga) : "memory");
}

__device__ __forceinline__ void wait_async0() {
  asm volatile("s_wait_asynccnt 0x0" ::: "memory");
}

// ---------------------------------------------------------------------------
// Convert emb (32000x256), W_iou (384x256), U_iou (384x256), U_f (256x256)
// to bf16.  emb_bf: 8192000 el, then W|U|Uf: 98304|98304|65536.
// ---------------------------------------------------------------------------
__global__ void __launch_bounds__(256)
cvt_kernel(const float* __restrict__ emb,
           const float* __restrict__ W_iou,
           const float* __restrict__ U_iou,
           const float* __restrict__ U_f,
           unsigned short* __restrict__ emb_bf,
           unsigned short* __restrict__ wbf)
{
  int idx = blockIdx.x * blockDim.x + threadIdx.x;   // grid covers 8454144
  if (idx < 8192000) {
    emb_bf[idx] = f2bf(emb[idx]);
  } else {
    int widx = idx - 8192000;
    float v;
    if (widx < 98304)       v = W_iou[widx];
    else if (widx < 196608) v = U_iou[widx - 98304];
    else                    v = U_f[widx - 196608];
    wbf[widx] = f2bf(v);
  }
}

// ---------------------------------------------------------------------------
// Leaf stage: 32 rows / block, 4 waves.  iou = emb_bf[leaf_x] @ W_iou^T;
// c = sig(i)*tanh(u); h = sig(o)*tanh(c).  Wave w owns cols [32w,32w+32) of
// each i/o/u segment, for two 16-row tiles (B fragments reused across both).
// ---------------------------------------------------------------------------
__global__ void __launch_bounds__(128)
leaf_kernel(const int* __restrict__ leaf_x,
            const unsigned short* __restrict__ emb_bf,   // [32000][256] bf16
            const unsigned short* __restrict__ Wbf,      // [384][256] bf16
            const float* __restrict__ b_iou,
            unsigned short* __restrict__ h_out,          // [NLEAF][128] bf16
            float* __restrict__ c_out)                   // [NLEAF][128] f32
{
  __shared__ __align__(16) unsigned short Abf[32][LDSPITCH];
  const int tid = threadIdx.x;
  const int m0  = blockIdx.x * 32;

  // Async-stage A tile: 4 threads per row, 8 x 16B chunks each.
  {
    const int r  = tid >> 2;              // 0..31
    const int kc = (tid & 3) * 64;        // bf16 col base
    const unsigned short* src = emb_bf + (size_t)leaf_x[m0 + r] * XDIM + kc;
    unsigned lds0 = (unsigned)(uintptr_t)(&Abf[r][kc]);
    #pragma unroll
    for (int j = 0; j < 8; ++j)
      async_b128(lds0 + 16u * j, src + 8 * j);
  }
  wait_async0();
  __syncthreads();

  const int lane = tid & 31;
  const int w    = tid >> 5;     // wave 0..3
  const int g    = lane >> 4;
  const int ln   = lane & 15;

  v8f zero = {};
  v8f acc[2][6];                 // [m-tile][i0,i1,o0,o1,u0,u1]
  #pragma unroll
  for (int mt = 0; mt < 2; ++mt)
    #pragma unroll
    for (int t = 0; t < 6; ++t) acc[mt][t] = zero;

  int nb[6];
  nb[0] = 32*w;        nb[1] = 32*w + 16;        // i
  nb[2] = 128 + nb[0]; nb[3] = 128 + nb[1];      // o
  nb[4] = 256 + nb[0]; nb[5] = 256 + nb[1];      // u

  #pragma unroll
  for (int kk = 0; kk < 8; ++kk) {               // K = 256, steps of 32
    const int ka = kk*32 + g*8;                  // A: two 16B chunks per lane
    ABPack a0, a1;
    a0.q[0] = *(const vi4*)(&Abf[ln][ka]);
    a0.q[1] = *(const vi4*)(&Abf[ln][ka + 16]);
    a1.q[0] = *(const vi4*)(&Abf[16 + ln][ka]);
    a1.q[1] = *(const vi4*)(&Abf[16 + ln][ka + 16]);
    const int kb = kk*32 + g*8;                  // B: two 16B chunks per lane
    #pragma unroll
    for (int t = 0; t < 6; ++t) {
      const unsigned short* brow = Wbf + (size_t)(nb[t] + ln) * XDIM + kb;
      ABPack b;
      b.q[0] = *(const vi4*)(brow);
      b.q[1] = *(const vi4*)(brow + 16);
      acc[0][t] = __builtin_amdgcn_wmma_f32_16x16x32_bf16(
          false, a0.v, false, b.v, (short)0, acc[0][t], false, false);
      acc[1][t] = __builtin_amdgcn_wmma_f32_16x16x32_bf16(
          false, a1.v, false, b.v, (short)0, acc[1][t], false, false);
    }
  }

  #pragma unroll
  for (int t = 0; t < 2; ++t) {
    const int col = 32*w + t*16 + ln;            // 0..127
    const float bi = b_iou[col];
    const float bo = b_iou[128 + col];
    const float bu = b_iou[256 + col];
    #pragma unroll
    for (int mt = 0; mt < 2; ++mt) {
      #pragma unroll
      for (int e = 0; e < 8; ++e) {              // C/D layout: row = e + 8*g
        const int row = m0 + mt*16 + e + 8*g;
        float iv = acc[mt][0 + t][e] + bi;
        float ov = acc[mt][2 + t][e] + bo;
        float uv = acc[mt][4 + t][e] + bu;
        float cc = sigf(iv) * tanhf(uv);
        float hh = sigf(ov) * tanhf(cc);
        h_out[(size_t)row * HDIM + col] = f2bf(hh);
        c_out[(size_t)row * HDIM + col] = cc;
      }
    }
  }
}

// ---------------------------------------------------------------------------
// Internal node stage: 32 rows / block.  h_cat = [h_prev[ci0], h_prev[ci1]];
// f = sig(h_cat @ U_f^T + b_f); csum = f0*c_prev[ci0] + f1*c_prev[ci1];
// iou = h_cat @ U_iou^T; c = sig(i)*tanh(u)+csum; h = sig(o)*tanh(c).
// Wave w owns cols [32w,32w+32) of i,o,u,f0,f1 for two row tiles.
// ---------------------------------------------------------------------------
__global__ void __launch_bounds__(128)
node_kernel(const int* __restrict__ ci,                  // [n][2]
            const unsigned short* __restrict__ h_prev,   // bf16 [prev][128]
            const float* __restrict__ c_prev,            // f32  [prev][128]
            const unsigned short* __restrict__ Ubf,      // U_iou [384][256]
            const unsigned short* __restrict__ Ufbf,     // U_f   [256][256]
            const float* __restrict__ b_iou,
            const float* __restrict__ b_f,
            unsigned short* __restrict__ h_out,
            float* __restrict__ c_out)
{
  __shared__ __align__(16) unsigned short Abf[32][LDSPITCH];
  __shared__ int ci0s[32];
  __shared__ int ci1s[32];
  const int tid = threadIdx.x;
  const int m0  = blockIdx.x * 32;

  {
    const int r = tid >> 2;                // 0..31
    const int q = tid & 3;                 // quarter: 0,1 -> child0; 2,3 -> child1
    const int c0 = ci[(m0 + r) * 2 + 0];
    const int c1 = ci[(m0 + r) * 2 + 1];
    if (q == 0) { ci0s[r] = c0; ci1s[r] = c1; }
    const int child = q >> 1;
    const int scol  = (q & 1) * 64;        // col within child's 128
    const unsigned short* src =
        h_prev + (size_t)(child ? c1 : c0) * HDIM + scol;
    unsigned lds0 = (unsigned)(uintptr_t)(&Abf[r][child * 128 + scol]);
    #pragma unroll
    for (int j = 0; j < 8; ++j)
      async_b128(lds0 + 16u * j, src + 8 * j);
  }
  wait_async0();
  __syncthreads();

  const int lane = tid & 31;
  const int w    = tid >> 5;
  const int g    = lane >> 4;
  const int ln   = lane & 15;

  v8f zero = {};
  v8f acc[2][10];          // [m-tile][i0,i1,o0,o1,u0,u1,f00,f01,f10,f11]
  #pragma unroll
  for (int mt = 0; mt < 2; ++mt)
    #pragma unroll
    for (int t = 0; t < 10; ++t) acc[mt][t] = zero;

  int nb[6];
  nb[0] = 32*w;        nb[1] = 32*w + 16;
  nb[2] = 128 + nb[0]; nb[3] = 128 + nb[1];
  nb[4] = 256 + nb[0]; nb[5] = 256 + nb[1];
  int nf[4];
  nf[0] = 32*w;        nf[1] = 32*w + 16;        // f child0
  nf[2] = 128 + nf[0]; nf[3] = 128 + nf[1];      // f child1

  #pragma unroll
  for (int kk = 0; kk < 8; ++kk) {
    const int ka = kk*32 + g*8;
    ABPack a0, a1;
    a0.q[0] = *(const vi4*)(&Abf[ln][ka]);
    a0.q[1] = *(const vi4*)(&Abf[ln][ka + 16]);
    a1.q[0] = *(const vi4*)(&Abf[16 + ln][ka]);
    a1.q[1] = *(const vi4*)(&Abf[16 + ln][ka + 16]);
    const int kb = kk*32 + g*8;
    #pragma unroll
    for (int t = 0; t < 6; ++t) {
      const unsigned short* brow = Ubf + (size_t)(nb[t] + ln) * XDIM + kb;
      ABPack b;
      b.q[0] = *(const vi4*)(brow);
      b.q[1] = *(const vi4*)(brow + 16);
      acc[0][t] = __builtin_amdgcn_wmma_f32_16x16x32_bf16(
          false, a0.v, false, b.v, (short)0, acc[0][t], false, false);
      acc[1][t] = __builtin_amdgcn_wmma_f32_16x16x32_bf16(
          false, a1.v, false, b.v, (short)0, acc[1][t], false, false);
    }
    #pragma unroll
    for (int t = 0; t < 4; ++t) {
      const unsigned short* brow = Ufbf + (size_t)(nf[t] + ln) * XDIM + kb;
      ABPack b;
      b.q[0] = *(const vi4*)(brow);
      b.q[1] = *(const vi4*)(brow + 16);
      acc[0][6 + t] = __builtin_amdgcn_wmma_f32_16x16x32_bf16(
          false, a0.v, false, b.v, (short)0, acc[0][6 + t], false, false);
      acc[1][6 + t] = __builtin_amdgcn_wmma_f32_16x16x32_bf16(
          false, a1.v, false, b.v, (short)0, acc[1][6 + t], false, false);
    }
  }

  #pragma unroll
  for (int t = 0; t < 2; ++t) {
    const int col = 32*w + t*16 + ln;            // 0..127
    const float bi  = b_iou[col];
    const float bo  = b_iou[128 + col];
    const float bu  = b_iou[256 + col];
    const float bf0 = b_f[col];
    const float bf1 = b_f[128 + col];
    #pragma unroll
    for (int mt = 0; mt < 2; ++mt) {
      #pragma unroll
      for (int e = 0; e < 8; ++e) {
        const int r  = mt*16 + e + 8*g;          // row within block
        const int i0 = ci0s[r];
        const int i1 = ci1s[r];
        float f0 = sigf(acc[mt][6 + t][e] + bf0);
        float f1 = sigf(acc[mt][8 + t][e] + bf1);
        float cs = f0 * c_prev[(size_t)i0 * HDIM + col]
                 + f1 * c_prev[(size_t)i1 * HDIM + col];
        float iv = acc[mt][0 + t][e] + bi;
        float ov = acc[mt][2 + t][e] + bo;
        float uv = acc[mt][4 + t][e] + bu;
        float cc = sigf(iv) * tanhf(uv) + cs;
        float hh = sigf(ov) * tanhf(cc);
        h_out[(size_t)(m0 + r) * HDIM + col] = f2bf(hh);
        c_out[(size_t)(m0 + r) * HDIM + col] = cc;
      }
    }
  }
}

// ---------------------------------------------------------------------------
// logits = h_all @ W_lin^T + b_lin   (128 -> 5); memory-bound, VALU.
// ---------------------------------------------------------------------------
__global__ void __launch_bounds__(256)
logits_kernel(const unsigned short* __restrict__ h_all,  // bf16
              const float* __restrict__ W_lin,           // [5][128]
              const float* __restrict__ b_lin,
              float* __restrict__ out, int rows)
{
  int row = blockIdx.x * blockDim.x + threadIdx.x;
  if (row >= rows) return;
  float a0 = b_lin[0], a1 = b_lin[1], a2 = b_lin[2], a3 = b_lin[3], a4 = b_lin[4];
  const unsigned short* h = h_all + (size_t)row * HDIM;
  #pragma unroll 4
  for (int k = 0; k < HDIM; ++k) {
    float hv = bf2f(h[k]);
    a0 += hv * W_lin[0*HDIM + k];
    a1 += hv * W_lin[1*HDIM + k];
    a2 += hv * W_lin[2*HDIM + k];
    a3 += hv * W_lin[3*HDIM + k];
    a4 += hv * W_lin[4*HDIM + k];
  }
  float* o = out + (size_t)row * 5;
  o[0] = a0; o[1] = a1; o[2] = a2; o[3] = a3; o[4] = a4;
}

// ---------------------------------------------------------------------------
extern "C" void kernel_launch(void* const* d_in, const int* in_sizes, int n_in,
                              void* d_out, int out_size, void* d_ws, size_t ws_size,
                              hipStream_t stream)
{
  (void)in_sizes; (void)n_in; (void)out_size; (void)ws_size;

  const int*   leaf_x = (const int*)  d_in[0];
  const int*   child  = (const int*)  d_in[1];
  const float* emb    = (const float*)d_in[2];
  const float* W_iou  = (const float*)d_in[3];
  const float* U_iou  = (const float*)d_in[4];
  const float* b_iou  = (const float*)d_in[5];
  const float* U_f    = (const float*)d_in[6];
  const float* b_f    = (const float*)d_in[7];
  const float* W_lin  = (const float*)d_in[8];
  const float* b_lin  = (const float*)d_in[9];

  const int NLEAF    = 262144;               // 32768 << 3
  const int ROWS_ALL = 491520;               // 262144+131072+65536+32768

  // Workspace: h_all(bf16) | c_all(f32) | emb_bf(bf16) | weights(bf16)
  unsigned short* h_all  = (unsigned short*)d_ws;
  float*          c_all  = (float*)(h_all + (size_t)ROWS_ALL * HDIM);
  unsigned short* emb_bf = (unsigned short*)(c_all + (size_t)ROWS_ALL * HDIM);
  unsigned short* wbf    = emb_bf + (size_t)32000 * XDIM;
  unsigned short* Wbf    = wbf;              // 98304
  unsigned short* Ubf    = wbf + 98304;      // 98304
  unsigned short* Ufbf   = wbf + 196608;     // 65536

  // 8192000 + 262144 = 8454144 elements -> 33024 blocks of 256.
  cvt_kernel<<<33024, 256, 0, stream>>>(emb, W_iou, U_iou, U_f, emb_bf, wbf);

  leaf_kernel<<<NLEAF / 32, 128, 0, stream>>>(leaf_x, emb_bf, Wbf, b_iou,
                                              h_all, c_all);

  const int lvl_off[4] = {0, 262144, 393216, 458752};
  int ci_off = 0;
  for (int l = 1; l <= 3; ++l) {
    const int n = NLEAF >> l;
    node_kernel<<<n / 32, 128, 0, stream>>>(
        child + (size_t)ci_off * 2,
        h_all + (size_t)lvl_off[l - 1] * HDIM,
        c_all + (size_t)lvl_off[l - 1] * HDIM,
        Ubf, Ufbf, b_iou, b_f,
        h_all + (size_t)lvl_off[l] * HDIM,
        c_all + (size_t)lvl_off[l] * HDIM);
    ci_off += n;
  }

  logits_kernel<<<(ROWS_ALL + 255) / 256, 256, 0, stream>>>(
      h_all, W_lin, b_lin, (float*)d_out, ROWS_ALL);
}
